// RPNHead_60593398612381
// MI455X (gfx1250) — matrix operand order repaired and checked by reading, not compiled, not run on hardware
//
#include <hip/hip_runtime.h>

// ---------------------------------------------------------------------------
// RPN head for MI455X (gfx1250): 3x3 conv 256->512 as implicit GEMM on
// v_wmma_f32_16x16x32_bf16 (f32 accumulate), M-tile=32 (8 WMMA / K-chunk /
// wave), fused bias+ReLU, 1x1 convs, softmax, concatenated scatter.
// ---------------------------------------------------------------------------

typedef __attribute__((ext_vector_type(16))) __bf16 v16bf;
typedef __attribute__((ext_vector_type(8)))  float  v8f;

#define FPN_C    256
#define SHARED_C 512
#define KTOT     2304          // 9 taps * 256 cin
#define TILE_M   32
#define NUM_A    261888        // anchors per batch (sum H*W*3)
#define B_STRIDE 40            // shorts per LDS B row (32 + pad, 16B aligned)
#define A_STRIDE 40            // shorts per LDS A row

__device__ __forceinline__ unsigned short f2bf(float f) {
    unsigned u = __float_as_uint(f);
    u += 0x7FFFu + ((u >> 16) & 1u);         // round-to-nearest-even
    return (unsigned short)(u >> 16);
}
__device__ __forceinline__ float bf2f(unsigned short s) {
    return __uint_as_float(((unsigned)s) << 16);
}

// --- Kernel 1: pack w_shared (3,3,256,512) fp32 HWIO -> bf16 Wp[n][k] -------
__global__ void pack_wshared_kernel(const float* __restrict__ w,
                                    unsigned short* __restrict__ wp) {
    int idx = blockIdx.x * 256 + threadIdx.x;     // idx = k*512 + n
    if (idx < KTOT * SHARED_C) {
        int k = idx >> 9;
        int n = idx & 511;
        wp[(size_t)n * KTOT + k] = f2bf(w[idx]);
    }
}

// --- LDS layout (union: GEMM tiles vs post-GEMM head) -----------------------
union alignas(16) SMem {
    struct {
        unsigned short B[SHARED_C * B_STRIDE];   // 40960 B
        unsigned short A[TILE_M * A_STRIDE];     //  2560 B
    } s1;                                        // 43520 B
    struct {
        unsigned short S[TILE_M * SHARED_C];     // 32768 B shared act (bf16)
        unsigned short W1[SHARED_C * 18];        // 18432 B 1x1 weights (bf16)
        float          O[TILE_M * 18];           //  2304 B 1x1 outputs
    } s2;                                        // 53504 B
};

// --- Kernel 2: fused RPN head per 32-pixel tile -----------------------------
__global__ __launch_bounds__(256)
void rpn_head_kernel(const float* __restrict__ f0, const float* __restrict__ f1,
                     const float* __restrict__ f2, const float* __restrict__ f3,
                     const float* __restrict__ f4,
                     const unsigned short* __restrict__ Wp,
                     const float* __restrict__ bsh,
                     const float* __restrict__ wcls, const float* __restrict__ bcls,
                     const float* __restrict__ wdel, const float* __restrict__ bdel,
                     float* __restrict__ out) {
    __shared__ SMem sm;
    const int tid  = threadIdx.x;
    const int lane = tid & 31;
    const int wave = tid >> 5;

    // ---- decode workgroup -> (batch, level, tile) ----
    const int wg = blockIdx.x;
    const int b  = wg / 2728;          // 32-pixel tiles per batch = 2728
    const int t  = wg % 2728;
    int li, tile;
    if      (t < 2048) { li = 0; tile = t;        }
    else if (t < 2560) { li = 1; tile = t - 2048; }
    else if (t < 2688) { li = 2; tile = t - 2560; }
    else if (t < 2720) { li = 3; tile = t - 2688; }
    else               { li = 4; tile = t - 2720; }
    const int logWs[5] = {8, 7, 6, 5, 4};
    const int aoffs[5] = {0, 196608, 245760, 258048, 261120};
    const int logW    = logWs[li];
    const int lvlAnch = aoffs[li];
    const int W = 1 << logW;
    const int H = W;
    const float* feat = (li == 0) ? f0 : (li == 1) ? f1 : (li == 2) ? f2
                        : (li == 3) ? f3 : f4;
    const int pix0 = tile * TILE_M;

    // Per-thread A-loader constants (fixed pixel & channel-quad per thread)
    const int pA  = tid >> 3;          // pixel row 0..31 handled by this thread
    const int cqA = (tid & 7) << 2;    // channel quad 0,4,...,28
    const int hA  = (pix0 + pA) >> logW;
    const int wA  = (pix0 + pA) & (W - 1);

    v8f acc[2][4] = {};   // per wave: 32 pixels x 64 out channels

    // ---- K loop: 9 taps x 8 chunks of 32 cin ----
    for (int tap = 0; tap < 9; ++tap) {
        const int dh = tap / 3 - 1;            // hoisted: 9x instead of 72x
        const int dw = tap % 3 - 1;
        const int hh = hA + dh, ww = wA + dw;
        const bool inb = ((unsigned)hh < (unsigned)H) &&
                         ((unsigned)ww < (unsigned)W);
        const float* srcBase = feat +
            ((((size_t)b * H + (inb ? hh : 0)) * W + (inb ? ww : 0)) * FPN_C + cqA);
        const size_t kTap = (size_t)tap * 256;

#pragma unroll 1
        for (int c8 = 0; c8 < 8; ++c8) {
            const int cin0 = c8 << 5;
            const size_t k0 = kTap + cin0;

            // A tile: 32 pixels x 32 cin, fp32 -> bf16, zero-padded at borders
            {
                float x0 = 0.f, x1 = 0.f, x2 = 0.f, x3 = 0.f;
                if (inb) {
                    float4 v = *(const float4*)(srcBase + cin0);
                    x0 = v.x; x1 = v.y; x2 = v.z; x3 = v.w;
                }
                uint2 pk;
                pk.x = (unsigned)f2bf(x0) | ((unsigned)f2bf(x1) << 16);
                pk.y = (unsigned)f2bf(x2) | ((unsigned)f2bf(x3) << 16);
                *(uint2*)&sm.s1.A[pA * A_STRIDE + cqA] = pk;
            }
            // B tile: 512 n x 32 k bf16 from packed weights (2 rows/thread)
            {
                const uint4* s0 = (const uint4*)(Wp + (size_t)tid * KTOT + k0);
                uint4* d0 = (uint4*)&sm.s1.B[tid * B_STRIDE];
                d0[0] = s0[0]; d0[1] = s0[1]; d0[2] = s0[2]; d0[3] = s0[3];
                const uint4* s1 =
                    (const uint4*)(Wp + (size_t)(tid + 256) * KTOT + k0);
                uint4* d1 = (uint4*)&sm.s1.B[(tid + 256) * B_STRIDE];
                d1[0] = s1[0]; d1[1] = s1[1]; d1[2] = s1[2]; d1[3] = s1[3];
                // near-cache prefetch of next chunk's weight rows
                __builtin_prefetch(Wp + (size_t)tid * KTOT + k0 + 32, 0, 3);
                __builtin_prefetch(Wp + (size_t)(tid + 256) * KTOT + k0 + 32, 0, 3);
            }
            __syncthreads();

            // A fragments (16-bit A 16x32 layout)
            union FA { v16bf v; uint4 q[2]; } fa[2];
            {
                const int kb = (lane >> 4) << 4;                // 0 or 16
#pragma unroll
                for (int mi = 0; mi < 2; ++mi) {
                    const int m = (mi << 4) + (lane & 15);
                    fa[mi].q[0] = *(const uint4*)&sm.s1.A[m * A_STRIDE + kb];
                    fa[mi].q[1] = *(const uint4*)&sm.s1.A[m * A_STRIDE + kb + 8];
                }
            }
            // B fragments: preload all 4 so ds loads overlap WMMA issue
            union FB { v16bf v; uint4 q[2]; } fb[4];
            {
                const int kbB = (lane >> 4) << 4;               // 0 or 16
#pragma unroll
                for (int ni = 0; ni < 4; ++ni) {
                    const int n = (wave << 6) + (ni << 4) + (lane & 15);
                    const uint4* pB =
                        (const uint4*)&sm.s1.B[n * B_STRIDE + kbB];
                    fb[ni].q[0] = pB[0]; fb[ni].q[1] = pB[1];
                }
            }
#pragma unroll
            for (int ni = 0; ni < 4; ++ni) {
#pragma unroll
                for (int mi = 0; mi < 2; ++mi) {
                    acc[mi][ni] = __builtin_amdgcn_wmma_f32_16x16x32_bf16(
                        false, fa[mi].v, false, fb[ni].v, (short)0,
                        acc[mi][ni], false, false);
                }
            }
            __syncthreads();
        }
    }

    // ---- bias + ReLU, stash shared activations as bf16 ----
#pragma unroll
    for (int ni = 0; ni < 4; ++ni) {
        const int n  = (wave << 6) + (ni << 4) + (lane & 15);
        const float bn = bsh[n];
        const int mB = (lane >> 4) << 3;
#pragma unroll
        for (int mi = 0; mi < 2; ++mi) {
#pragma unroll
            for (int r = 0; r < 8; ++r) {
                float v = fmaxf(acc[mi][ni][r] + bn, 0.f);
                sm.s2.S[((mi << 4) + mB + r) * SHARED_C + n] = f2bf(v);
            }
        }
    }
    // ---- preload 1x1 weights into LDS: W1[n][0..5]=cls, [6..17]=delta ----
    for (int idx = tid; idx < SHARED_C * 6; idx += 256) {
        sm.s2.W1[(idx / 6) * 18 + (idx % 6)] = f2bf(wcls[idx]);
    }
    for (int idx = tid; idx < SHARED_C * 12; idx += 256) {
        sm.s2.W1[(idx / 12) * 18 + 6 + (idx % 12)] = f2bf(wdel[idx]);
    }
    __syncthreads();

    // ---- 1x1 convs: 32 pixels x 18 outputs, dot over 512 channels ----
    for (int idx = tid; idx < TILE_M * 18; idx += 256) {
        const int m = idx / 18, j = idx % 18;
        float a = (j < 6) ? bcls[j] : bdel[j - 6];
        const unsigned short* Srow = &sm.s2.S[m * SHARED_C];
#pragma unroll 8
        for (int n = 0; n < SHARED_C; ++n)
            a += bf2f(Srow[n]) * bf2f(sm.s2.W1[n * 18 + j]);
        sm.s2.O[m * 18 + j] = a;
    }
    __syncthreads();

    // ---- write logits / probs / deltas (concatenated flat output) ----
    float* outLog  = out;
    float* outProb = out + 1047552;      // 2*NUM_A*2
    float* outDel  = out + 2095104;      // 2*(2*NUM_A*2)
    for (int idx = tid; idx < TILE_M * 3; idx += 256) {   // logits + softmax
        const int m = idx / 3, r = idx % 3;
        const float l0 = sm.s2.O[m * 18 + 2 * r];
        const float l1 = sm.s2.O[m * 18 + 2 * r + 1];
        const long long a  = (long long)lvlAnch + (long long)(pix0 + m) * 3 + r;
        const long long gi = (long long)b * NUM_A + a;
        outLog[gi * 2 + 0] = l0;
        outLog[gi * 2 + 1] = l1;
        const float mx = fmaxf(l0, l1);
        const float e0 = __expf(l0 - mx);
        const float e1 = __expf(l1 - mx);
        const float inv = 1.f / (e0 + e1);
        outProb[gi * 2 + 0] = e0 * inv;
        outProb[gi * 2 + 1] = e1 * inv;
    }
    for (int idx = tid; idx < TILE_M * 12; idx += 256) {  // deltas
        const int m = idx / 12, j = idx % 12;
        const int r = j >> 2, c = j & 3;
        const long long a  = (long long)lvlAnch + (long long)(pix0 + m) * 3 + r;
        const long long gi = (long long)b * NUM_A + a;
        outDel[gi * 4 + c] = sm.s2.O[m * 18 + 6 + j];
    }
}

// ---------------------------------------------------------------------------
extern "C" void kernel_launch(void* const* d_in, const int* in_sizes, int n_in,
                              void* d_out, int out_size, void* d_ws, size_t ws_size,
                              hipStream_t stream) {
    (void)in_sizes; (void)n_in; (void)out_size; (void)ws_size;
    const float* f0   = (const float*)d_in[0];
    const float* f1   = (const float*)d_in[1];
    const float* f2   = (const float*)d_in[2];
    const float* f3   = (const float*)d_in[3];
    const float* f4   = (const float*)d_in[4];
    const float* wsh  = (const float*)d_in[5];
    const float* bsh  = (const float*)d_in[6];
    const float* wcls = (const float*)d_in[7];
    const float* bcls = (const float*)d_in[8];
    const float* wdel = (const float*)d_in[9];
    const float* bdel = (const float*)d_in[10];
    unsigned short* Wp = (unsigned short*)d_ws;   // 512*2304 bf16 = 2.25 MB

    pack_wshared_kernel<<<(KTOT * SHARED_C + 255) / 256, 256, 0, stream>>>(wsh, Wp);
    rpn_head_kernel<<<5456, 256, 0, stream>>>(f0, f1, f2, f3, f4, Wp, bsh,
                                              wcls, bcls, wdel, bdel,
                                              (float*)d_out);
}
